// PlanTokenizer_19439021982142
// MI455X (gfx1250) — matrix operand order repaired and checked by reading, not compiled, not run on hardware
//
#include <hip/hip_runtime.h>

typedef __attribute__((ext_vector_type(16))) __bf16        v16bf;
typedef __attribute__((ext_vector_type(8)))  float         v8f;
typedef __attribute__((ext_vector_type(4)))  unsigned int  u32x4;
typedef __attribute__((ext_vector_type(4)))  float         f32x4;
typedef int v4i __attribute__((vector_size(16)));
typedef __attribute__((address_space(1))) v4i* gptr_v4i;
typedef __attribute__((address_space(3))) v4i* lptr_v4i;

#define P_DIM 512
#define C_DIM 256
#define K_CB  128
#define TILE_M 128
#define NT1 16           /* 256/16 col tiles in GEMM1 */
#define NT2 8            /* 128/16 code tiles in GEMM2 */
#define KSTEP 32
#define NKS1 (P_DIM/KSTEP)   /* 16 */
#define NKS2 (C_DIM/KSTEP)   /* 8 */

/* LDS strides in bf16 elements, padded so ds_load_b128 lanes hit distinct banks */
#define A_STRIDE 40
#define W_STRIDE 40
#define Z_STRIDE 264
#define CB_STRIDE 264

#define A_BUF_BYTES (TILE_M*A_STRIDE*2)          /* 10240 */
#define W_BUF_BYTES (C_DIM*W_STRIDE*2)           /* 20480 */
#define OFF_A0 0
#define OFF_A1 (OFF_A0 + A_BUF_BYTES)            /* 10240 */
#define OFF_W0 (OFF_A1 + A_BUF_BYTES)            /* 20480 */
#define OFF_W1 (OFF_W0 + W_BUF_BYTES)            /* 40960, ends 61440 */
#define OFF_ZT 0                                 /* reuses phase-1 buffers */
#define OFF_CB (TILE_M*Z_STRIDE*2)               /* 67584 (disjoint from phase-1 bufs) */
#define OFF_ROWIDX (OFF_CB + K_CB*CB_STRIDE*2)   /* 135168 */
#define OFF_CBNORM (OFF_ROWIDX + TILE_M*4)       /* 135680 */
#define SMEM_BYTES (OFF_CBNORM + K_CB*4)         /* 136192 */

/* ---- gfx1250 async global->LDS (guarded; falls back to load+store) ---- */
#if defined(__has_builtin)
#  if __has_builtin(__builtin_amdgcn_global_load_async_to_lds_b128)
#    define HAVE_ASYNC_LDS 1
#  endif
#endif
#ifndef HAVE_ASYNC_LDS
#  define HAVE_ASYNC_LDS 0
#endif

__device__ __forceinline__ void async_copy16(const unsigned short* g, unsigned short* l) {
#if HAVE_ASYNC_LDS
  __builtin_amdgcn_global_load_async_to_lds_b128(
      (gptr_v4i)(void*)g, (lptr_v4i)(void*)l, 0, 0);
#else
  *(u32x4*)(void*)l = *(const u32x4*)(const void*)g;
#endif
}
__device__ __forceinline__ void async_wait() {
#if HAVE_ASYNC_LDS
#  if __has_builtin(__builtin_amdgcn_s_wait_asynccnt)
  __builtin_amdgcn_s_wait_asynccnt(0);
#  else
  asm volatile("s_wait_asynccnt 0" ::: "memory");
#  endif
#endif
}

__device__ __forceinline__ unsigned pack2(float a, float b) {
  union { __bf16 h[2]; unsigned u; } v;
  v.h[0] = (__bf16)a; v.h[1] = (__bf16)b;   /* native v_cvt, RTNE */
  return v.u;
}
__device__ __forceinline__ float bf2f(unsigned h) {
  union { unsigned u; float f; } v; v.u = h << 16; return v.f;
}

union Frag { v16bf v; u32x4 q[2]; };

/* ---- prep: W,codebook -> bf16 workspace, codebook norms, zero loss ---- */
__global__ __launch_bounds__(256)
void prep_kernel(const float* __restrict__ W, const float* __restrict__ cb,
                 unsigned* __restrict__ wBF, unsigned* __restrict__ cbBF,
                 float* __restrict__ cbNorm, float* __restrict__ loss_out)
{
  const int t = threadIdx.x;
  const int b = blockIdx.x;
  if (b == 0 && t == 0) *loss_out = 0.0f;
  if (b < C_DIM) {                                  /* 256 W rows */
    const float* src = W + (long)b * P_DIM;
    wBF[b * (P_DIM / 2) + t] = pack2(src[2 * t], src[2 * t + 1]);
  } else {                                          /* 128 codebook rows */
    const int c = b - C_DIM;
    __shared__ float red[256];
    float p = 0.0f;
    if (t < 128) {
      const float a0 = cb[c * C_DIM + 2 * t], a1 = cb[c * C_DIM + 2 * t + 1];
      cbBF[c * (C_DIM / 2) + t] = pack2(a0, a1);
      p = a0 * a0 + a1 * a1;
    }
    red[t] = p;
    __syncthreads();
    if (t == 0) { float s = 0.0f; for (int i = 0; i < 128; ++i) s += red[i]; cbNorm[c] = s; }
  }
}

__device__ __forceinline__ void stage_w_async(const unsigned short* wBF,
                                              unsigned short* ldsBase, int tid, int k0) {
  const unsigned short* g = wBF + (long)tid * P_DIM + k0;
  unsigned short* l = ldsBase + tid * W_STRIDE;
  #pragma unroll
  for (int i = 0; i < 4; ++i) async_copy16(g + 8 * i, l + 8 * i);
}

__global__ __launch_bounds__(256)
void vq_fused_kernel(const float* __restrict__ plan,
                     const unsigned short* __restrict__ wBF,
                     const float* __restrict__ bias,
                     const float* __restrict__ cb,          /* fp32, for z_q gather */
                     const unsigned short* __restrict__ cbBF,
                     const float* __restrict__ cbNormG,
                     float* __restrict__ zq, float* __restrict__ idx_out,
                     float* __restrict__ loss_out, float loss_scale)
{
  extern __shared__ char smem[];
  unsigned short* zT  = (unsigned short*)(smem + OFF_ZT);
  unsigned short* cbS = (unsigned short*)(smem + OFF_CB);
  int*   rowIdx  = (int*)(smem + OFF_ROWIDX);
  float* cbNormS = (float*)(smem + OFF_CBNORM);

  const int tid  = threadIdx.x;
  const int lane = tid & 31;
  const int wave = tid >> 5;
  const int col0 = lane & 15;
  const int hi   = lane >> 4;
  const long tokBase = (long)blockIdx.x * TILE_M;

  /* codebook tile: pure bf16 copy -> async DMA, overlaps the phase-1 prologue */
  {
    const int code = tid & (K_CB - 1);
    const int half = tid >> 7;
    const unsigned short* g = cbBF + code * C_DIM + half * 128;
    unsigned short* l = cbS + code * CB_STRIDE + half * 128;
    #pragma unroll
    for (int i = 0; i < 16; ++i) async_copy16(g + 8 * i, l + 8 * i);
  }
  if (tid < K_CB) cbNormS[tid] = cbNormG[tid];

  /* ---- phase 1: z = plan * W^T  (M=128, N=256, K=512), pipelined ---- */
  v8f vzero = {0, 0, 0, 0, 0, 0, 0, 0};
  v8f acc[NT1];
  #pragma unroll
  for (int i = 0; i < NT1; ++i) acc[i] = vzero;

  f32x4 pre[8];
  /* prologue: stage ks=0 into buffer 0 */
  stage_w_async(wBF, (unsigned short*)(smem + OFF_W0), tid, 0);
  if (tid < TILE_M) {
    const f32x4* src = (const f32x4*)(plan + (tokBase + tid) * P_DIM);
    #pragma unroll
    for (int i = 0; i < 8; ++i) pre[i] = src[i];
    u32x4* d = (u32x4*)(void*)((unsigned short*)(smem + OFF_A0) + tid * A_STRIDE);
    #pragma unroll
    for (int i = 0; i < 4; ++i) {
      u32x4 q;
      q.x = pack2(pre[2*i].x,   pre[2*i].y);
      q.y = pack2(pre[2*i].z,   pre[2*i].w);
      q.z = pack2(pre[2*i+1].x, pre[2*i+1].y);
      q.w = pack2(pre[2*i+1].z, pre[2*i+1].w);
      d[i] = q;
    }
  }
  async_wait();
  __syncthreads();

  for (int ks = 0; ks < NKS1; ++ks) {
    const int cur = ks & 1;
    unsigned short* aCur = (unsigned short*)(smem + (cur ? OFF_A1 : OFF_A0));
    unsigned short* wCur = (unsigned short*)(smem + (cur ? OFF_W1 : OFF_W0));
    unsigned short* aNxt = (unsigned short*)(smem + (cur ? OFF_A0 : OFF_A1));
    unsigned short* wNxt = (unsigned short*)(smem + (cur ? OFF_W0 : OFF_W1));
    const bool more = (ks + 1 < NKS1);

    /* issue next-step global traffic first: async W copy + raw fp32 plan loads */
    if (more) {
      stage_w_async(wBF, wNxt, tid, (ks + 1) * KSTEP);
      if (tid < TILE_M) {
        const f32x4* src = (const f32x4*)(plan + (tokBase + tid) * P_DIM + (ks + 1) * KSTEP);
        #pragma unroll
        for (int i = 0; i < 8; ++i) pre[i] = src[i];
      }
    }

    /* A frag (contiguous, matches B): lane<16 -> K {0..15}, else {16..31} */
    Frag a;
    const unsigned short* ap = aCur + (wave * 16 + col0) * A_STRIDE + hi * 16;
    a.q[0] = *(const u32x4*)(const void*)(ap);
    a.q[1] = *(const u32x4*)(const void*)(ap + 8);
    /* B frags with 2-deep rotation so ds_loads overlap WMMA */
    Frag bc, bn;
    {
      const unsigned short* bp = wCur + col0 * W_STRIDE + hi * 16;
      bc.q[0] = *(const u32x4*)(const void*)(bp);
      bc.q[1] = *(const u32x4*)(const void*)(bp + 8);
    }
    #pragma unroll
    for (int nt = 0; nt < NT1; ++nt) {
      if (nt + 1 < NT1) {
        const unsigned short* bp = wCur + ((nt + 1) * 16 + col0) * W_STRIDE + hi * 16;
        bn.q[0] = *(const u32x4*)(const void*)(bp);
        bn.q[1] = *(const u32x4*)(const void*)(bp + 8);
      }
      acc[nt] = __builtin_amdgcn_wmma_f32_16x16x32_bf16(
          false, a.v, false, bc.v, (short)0, acc[nt], false, false);
      bc = bn;
    }

    /* convert+store next plan slab after WMMAs (loads have landed by now) */
    if (more && tid < TILE_M) {
      u32x4* d = (u32x4*)(void*)(aNxt + tid * A_STRIDE);
      #pragma unroll
      for (int i = 0; i < 4; ++i) {
        u32x4 q;
        q.x = pack2(pre[2*i].x,   pre[2*i].y);
        q.y = pack2(pre[2*i].z,   pre[2*i].w);
        q.z = pack2(pre[2*i+1].x, pre[2*i+1].y);
        q.w = pack2(pre[2*i+1].z, pre[2*i+1].w);
        d[i] = q;
      }
    }
    async_wait();
    __syncthreads();
  }

  /* ---- bias + spill z tile to LDS bf16 (reuses phase-1 buffers; barrier above) */
  {
    const int mbase = wave * 16 + hi * 8;
    #pragma unroll
    for (int nt = 0; nt < NT1; ++nt) {
      const float bv = bias[nt * 16 + col0];
      #pragma unroll
      for (int j = 0; j < 8; ++j)
        zT[(mbase + j) * Z_STRIDE + nt * 16 + col0] = (unsigned short)(pack2(acc[nt][j] + bv, 0.0f) & 0xffffu);
    }
  }
  __syncthreads();

  /* ---- phase 2: scores = z * cb^T (M=128, N=128, K=256) ---- */
  v8f acc2[NT2];
  #pragma unroll
  for (int i = 0; i < NT2; ++i) acc2[i] = vzero;

  for (int ks = 0; ks < NKS2; ++ks) {
    const int k0 = ks * KSTEP;
    Frag a;
    const unsigned short* ap = zT + (wave * 16 + col0) * Z_STRIDE + k0 + hi * 16;
    a.q[0] = *(const u32x4*)(const void*)(ap);
    a.q[1] = *(const u32x4*)(const void*)(ap + 8);
    Frag bc, bn;
    {
      const unsigned short* bp = cbS + col0 * CB_STRIDE + k0 + hi * 16;
      bc.q[0] = *(const u32x4*)(const void*)(bp);
      bc.q[1] = *(const u32x4*)(const void*)(bp + 8);
    }
    #pragma unroll
    for (int ct = 0; ct < NT2; ++ct) {
      if (ct + 1 < NT2) {
        const unsigned short* bp = cbS + ((ct + 1) * 16 + col0) * CB_STRIDE + k0 + hi * 16;
        bn.q[0] = *(const u32x4*)(const void*)(bp);
        bn.q[1] = *(const u32x4*)(const void*)(bp + 8);
      }
      acc2[ct] = __builtin_amdgcn_wmma_f32_16x16x32_bf16(
          false, a.v, false, bc.v, (short)0, acc2[ct], false, false);
      bc = bn;
    }
  }

  /* ---- per-row argmin of ||e||^2 - 2*dot ---- */
  float best[8]; int bidx[8];
  #pragma unroll
  for (int j = 0; j < 8; ++j) { best[j] = 3.4e38f; bidx[j] = 0; }
  #pragma unroll
  for (int ct = 0; ct < NT2; ++ct) {
    const float nrm = cbNormS[ct * 16 + col0];
    #pragma unroll
    for (int j = 0; j < 8; ++j) {
      const float s = nrm - 2.0f * acc2[ct][j];
      if (s < best[j]) { best[j] = s; bidx[j] = ct * 16 + col0; }
    }
  }
  #pragma unroll
  for (int m = 1; m < 16; m <<= 1) {
    #pragma unroll
    for (int j = 0; j < 8; ++j) {
      const float os = __shfl_xor(best[j], m, 32);
      const int   oi = __shfl_xor(bidx[j], m, 32);
      if (os < best[j] || (os == best[j] && oi < bidx[j])) { best[j] = os; bidx[j] = oi; }
    }
  }
  if (col0 == 0) {
    const int mbase = wave * 16 + hi * 8;
    #pragma unroll
    for (int j = 0; j < 8; ++j) rowIdx[mbase + j] = bidx[j];
  }
  __syncthreads();

  /* ---- outputs: z_q gather (fp32 codebook via L2), indices, commit loss ---- */
  if (tid < TILE_M) {
    const int r = tid;
    const int idx = rowIdx[r];
    const long tok = tokBase + r;
    idx_out[tok] = (float)idx;
    const f32x4* crow = (const f32x4*)(cb + (long)idx * C_DIM);
    f32x4* orow = (f32x4*)(zq + tok * C_DIM);
    const unsigned* zrow = (const unsigned*)(const void*)(zT + r * Z_STRIDE);
    float lsum = 0.0f;
    #pragma unroll 4
    for (int i = 0; i < 64; ++i) {
      f32x4 c4 = crow[i];
      const unsigned p0 = zrow[2 * i], p1 = zrow[2 * i + 1];
      const float d0 = c4.x - bf2f(p0 & 0xffffu);
      const float d1 = c4.y - bf2f(p0 >> 16);
      const float d2 = c4.z - bf2f(p1 & 0xffffu);
      const float d3 = c4.w - bf2f(p1 >> 16);
      lsum += d0 * d0 + d1 * d1 + d2 * d2 + d3 * d3;
      orow[i] = c4;
    }
    atomicAdd(loss_out, lsum * loss_scale);
  }
}

extern "C" void kernel_launch(void* const* d_in, const int* in_sizes, int n_in,
                              void* d_out, int out_size, void* d_ws, size_t ws_size,
                              hipStream_t stream)
{
  const float* plan = (const float*)d_in[0];
  const float* W    = (const float*)d_in[1];
  const float* b    = (const float*)d_in[2];
  const float* cb   = (const float*)d_in[3];
  const long ntok = (long)in_sizes[0] / P_DIM;          /* 131072 */

  float* zq       = (float*)d_out;                      /* [ntok*256] */
  float* idx_out  = zq + ntok * C_DIM;                  /* [ntok] */
  float* loss_out = idx_out + ntok;                     /* [1] */

  /* workspace: wBF (256KB) | cbBF (64KB) | cbNorm (512B) */
  unsigned short* wBF  = (unsigned short*)d_ws;
  unsigned short* cbBF = (unsigned short*)((char*)d_ws + (size_t)C_DIM * P_DIM * 2);
  float* cbNorm = (float*)((char*)d_ws + (size_t)C_DIM * P_DIM * 2 + (size_t)K_CB * C_DIM * 2);

  prep_kernel<<<C_DIM + K_CB, 256, 0, stream>>>(W, cb, (unsigned*)wBF, (unsigned*)cbBF,
                                                cbNorm, loss_out);

  const float scale = 1.25f / ((float)ntok * (float)C_DIM);
  vq_fused_kernel<<<(int)(ntok / TILE_M), 256, SMEM_BYTES, stream>>>(
      plan, wBF, b, cb, cbBF, cbNorm, zq, idx_out, loss_out, scale);
}
